// MultiLayerGCN_59115929862863
// MI455X (gfx1250) — compile-verified, matrix operand order checked by reading, not run on hardware
//
#include <hip/hip_runtime.h>
#include <hip/hip_bf16.h>

#define N_NODES   100000
#define N_EDGES   1600000
#define HID       128
#define OUT_CH    64
#define NUM_GRAPHS 64

typedef __attribute__((ext_vector_type(16))) _Float16 v16h;
typedef __attribute__((ext_vector_type(8)))  float    v8f;

// ---------- degree / normalization ----------
__global__ void k_init_deg(float* __restrict__ deg) {
    int i = blockIdx.x * blockDim.x + threadIdx.x;
    if (i < N_NODES) deg[i] = 1.0f;                    // self-loop
}

__global__ void k_accum_deg(const int* __restrict__ dst, float* __restrict__ deg) {
    int e = blockIdx.x * blockDim.x + threadIdx.x;
    if (e < N_EDGES) atomicAdd(&deg[dst[e]], 1.0f);
}

__global__ void k_dinv(float* __restrict__ deg) {
    int i = blockIdx.x * blockDim.x + threadIdx.x;
    if (i < N_NODES) deg[i] = rsqrtf(deg[i]);
}

__global__ void k_zero_pool(float* __restrict__ sums, float* __restrict__ counts) {
    int i = blockIdx.x * blockDim.x + threadIdx.x;
    if (i < NUM_GRAPHS * HID) sums[i] = 0.0f;
    if (i < NUM_GRAPHS)       counts[i] = 0.0f;
}

// ---------- WMMA GEMM: Y[N,128] = act(X)[N,128] @ W[128,128] + b ----------
__global__ __launch_bounds__(256)
void k_gemm_wmma(const float* __restrict__ X, const float* __restrict__ W,
                 const float* __restrict__ bias, float* __restrict__ Y,
                 int applyRelu) {
    __shared__ _Float16 Xs[128 * 128];   // Xs[m][k], row-major in k
    __shared__ _Float16 Wt[128 * 128];   // Wt[n][k] = W[k][n] (transposed)

    const int  tid      = threadIdx.x;
    const long rowBlock = (long)blockIdx.x * 128;

    // Cooperative staging: 16384 elements each of X-tile and W, f32 -> f16
    #pragma unroll 4
    for (int i = 0; i < 64; ++i) {
        int  idx = tid + i * 256;
        int  r   = idx >> 7;          // row within tile / k for W
        int  c   = idx & 127;         // col / n for W
        long grow = rowBlock + r;
        float xv = (grow < N_NODES) ? X[grow * 128 + c] : 0.0f;
        if (applyRelu) xv = fmaxf(xv, 0.0f);
        Xs[idx] = (_Float16)xv;
        Wt[c * 128 + r] = (_Float16)W[idx];   // transpose into LDS
    }
    __syncthreads();

    const int wv   = tid >> 5;        // wave id 0..7 -> 16-row strip
    const int lane = tid & 31;
    const int lo   = lane & 15;
    const int hi   = lane >> 4;
    const int m    = (wv << 4) + lo;  // A-matrix row index for this lane

    v8f acc[8] = {};

    #pragma unroll
    for (int k0 = 0; k0 < 128; k0 += 32) {
        // A: 16x32 f16, lane layout per CDNA5 ISA (two 16B contiguous chunks)
        v16h a;
        const _Float16* pa = &Xs[m * 128 + k0 + hi * 8];
        #pragma unroll
        for (int j = 0; j < 8; ++j) { a[j] = pa[j]; a[j + 8] = pa[j + 16]; }

        #pragma unroll
        for (int ct = 0; ct < 8; ++ct) {
            v16h b;
            const _Float16* pb = &Wt[(ct * 16 + lo) * 128 + k0 + hi * 8];
            #pragma unroll
            for (int j = 0; j < 8; ++j) { b[j] = pb[j]; b[j + 8] = pb[j + 16]; }
            acc[ct] = __builtin_amdgcn_wmma_f32_16x16x32_f16(
                false, a, false, b, (short)0, acc[ct], false, false);
        }
    }

    // D layout: lane -> N = lo, row M = r + 8*hi
    const long row0 = rowBlock + (wv << 4);
    #pragma unroll
    for (int ct = 0; ct < 8; ++ct) {
        int   col = ct * 16 + lo;
        float bb  = bias[col];
        #pragma unroll
        for (int r = 0; r < 8; ++r) {
            long row = row0 + r + 8 * hi;
            if (row < N_NODES) Y[row * 128 + col] = acc[ct][r] + bb;
        }
    }
}

// ---------- self-loop term: agg = h * dinv^2 ----------
__global__ void k_selfloop(const float* __restrict__ h, const float* __restrict__ dinv,
                           float* __restrict__ agg) {
    long idx = (long)blockIdx.x * blockDim.x + threadIdx.x;
    if (idx < (long)N_NODES * HID) {
        int   node = (int)(idx >> 7);
        float d    = dinv[node];
        agg[idx] = h[idx] * d * d;
    }
}

// ---------- edge scatter: one wave per edge, float4 per lane ----------
__global__ void k_scatter(const float* __restrict__ h, const int* __restrict__ src,
                          const int* __restrict__ dst, const float* __restrict__ dinv,
                          float* __restrict__ agg) {
    long tid = (long)blockIdx.x * blockDim.x + threadIdx.x;
    int  e   = (int)(tid >> 5);
    if (e >= N_EDGES) return;
    int   lane = (int)(tid & 31);
    int   s    = src[e];
    int   d    = dst[e];
    float nrm  = dinv[s] * dinv[d];
    float4 v = ((const float4*)(h + (long)s * 128))[lane];
    float* ap = agg + (long)d * 128 + lane * 4;
    atomicAdd(ap + 0, v.x * nrm);
    atomicAdd(ap + 1, v.y * nrm);
    atomicAdd(ap + 2, v.z * nrm);
    atomicAdd(ap + 3, v.w * nrm);
}

// ---------- mean-pool accumulation: one wave per node ----------
__global__ void k_pool(const float* __restrict__ h, const int* __restrict__ batch,
                       float* __restrict__ sums, float* __restrict__ counts) {
    long tid  = (long)blockIdx.x * blockDim.x + threadIdx.x;
    int  node = (int)(tid >> 5);
    if (node >= N_NODES) return;
    int lane = (int)(tid & 31);
    int g    = batch[node];
    float4 v = ((const float4*)(h + (long)node * 128))[lane];
    float* sp = sums + g * 128 + lane * 4;
    atomicAdd(sp + 0, v.x);
    atomicAdd(sp + 1, v.y);
    atomicAdd(sp + 2, v.z);
    atomicAdd(sp + 3, v.w);
    if (lane == 0) atomicAdd(&counts[g], 1.0f);
}

// ---------- final projection: out[g,o] = (sums[g,:]/cnt) @ Wp + bp ----------
__global__ void k_final(const float* __restrict__ sums, const float* __restrict__ counts,
                        const float* __restrict__ Wp, const float* __restrict__ bp,
                        float* __restrict__ out) {
    int tid = blockIdx.x * blockDim.x + threadIdx.x;
    if (tid >= NUM_GRAPHS * OUT_CH) return;
    int g = tid >> 6;
    int o = tid & 63;
    float inv = 1.0f / fmaxf(counts[g], 1.0f);
    float a = 0.0f;
    #pragma unroll 8
    for (int k = 0; k < HID; ++k) a += sums[g * 128 + k] * Wp[k * 64 + o];
    out[tid] = a * inv + bp[o];
}

extern "C" void kernel_launch(void* const* d_in, const int* in_sizes, int n_in,
                              void* d_out, int out_size, void* d_ws, size_t ws_size,
                              hipStream_t stream) {
    (void)in_sizes; (void)n_in; (void)out_size; (void)ws_size;
    const float* x     = (const float*)d_in[0];
    const int*   ei    = (const int*)d_in[1];
    const int*   src   = ei;
    const int*   dst   = ei + N_EDGES;
    const int*   batch = (const int*)d_in[2];
    const float* W1 = (const float*)d_in[3]; const float* b1 = (const float*)d_in[4];
    const float* W2 = (const float*)d_in[5]; const float* b2 = (const float*)d_in[6];
    const float* W3 = (const float*)d_in[7]; const float* b3 = (const float*)d_in[8];
    const float* Wp = (const float*)d_in[9]; const float* bp = (const float*)d_in[10];
    float* out = (float*)d_out;

    char*  ws  = (char*)d_ws;
    size_t off = 0;
    float* dinv   = (float*)(ws + off); off += ((size_t)N_NODES * 4 + 255) & ~(size_t)255;
    float* h1     = (float*)(ws + off); off += (size_t)N_NODES * HID * 4;
    float* h2     = (float*)(ws + off); off += (size_t)N_NODES * HID * 4;
    float* sums   = (float*)(ws + off); off += (size_t)NUM_GRAPHS * HID * 4;
    float* counts = (float*)(ws + off); off += (size_t)NUM_GRAPHS * 4;

    const dim3 B(256);
    const int  nodeBlocks = (N_NODES + 255) / 256;
    const int  edgeBlocks = (N_EDGES + 255) / 256;
    const int  gemmBlocks = (N_NODES + 127) / 128;
    const int  selfBlocks = (int)(((long)N_NODES * HID + 255) / 256);
    const int  scatBlocks = (int)(((long)N_EDGES * 32 + 255) / 256);
    const int  poolBlocks = (int)(((long)N_NODES * 32 + 255) / 256);

    k_init_deg <<<nodeBlocks, B, 0, stream>>>(dinv);
    k_accum_deg<<<edgeBlocks, B, 0, stream>>>(dst, dinv);
    k_dinv     <<<nodeBlocks, B, 0, stream>>>(dinv);
    k_zero_pool<<<(NUM_GRAPHS * HID + 255) / 256, B, 0, stream>>>(sums, counts);

    // layer 1: h1 = x@W1+b1 ; h2 = self + scatter
    k_gemm_wmma<<<gemmBlocks, B, 0, stream>>>(x,  W1, b1, h1, 0);
    k_selfloop <<<selfBlocks, B, 0, stream>>>(h1, dinv, h2);
    k_scatter  <<<scatBlocks, B, 0, stream>>>(h1, src, dst, dinv, h2);
    // layer 2 (relu fused into GEMM input staging)
    k_gemm_wmma<<<gemmBlocks, B, 0, stream>>>(h2, W2, b2, h1, 1);
    k_selfloop <<<selfBlocks, B, 0, stream>>>(h1, dinv, h2);
    k_scatter  <<<scatBlocks, B, 0, stream>>>(h1, src, dst, dinv, h2);
    // layer 3 (no relu after)
    k_gemm_wmma<<<gemmBlocks, B, 0, stream>>>(h2, W3, b3, h1, 1);
    k_selfloop <<<selfBlocks, B, 0, stream>>>(h1, dinv, h2);
    k_scatter  <<<scatBlocks, B, 0, stream>>>(h1, src, dst, dinv, h2);

    k_pool <<<poolBlocks, B, 0, stream>>>(h2, batch, sums, counts);
    k_final<<<(NUM_GRAPHS * OUT_CH + 255) / 256, B, 0, stream>>>(sums, counts, Wp, bp, out);
}
